// GLANaive_69982197121471
// MI455X (gfx1250) — matrix operand order repaired and checked by reading, not compiled, not run on hardware
//
#include <hip/hip_runtime.h>

typedef __attribute__((ext_vector_type(16))) _Float16 v16h;
typedef __attribute__((ext_vector_type(8)))  _Float16 v8h;
typedef __attribute__((ext_vector_type(4)))  _Float16 v4h;
typedef __attribute__((ext_vector_type(8)))  float    v8f;
typedef __attribute__((ext_vector_type(4)))  float    v4f;

#define D_    512
#define L_    16384
#define B_    2
#define BL_   32768
#define CH_   64
#define NCH_  256

union V16U { v16h v; v8h h[2]; };

__device__ __forceinline__ v8f wmma16(v16h a, v16h b, v8f c) {
  // D = A(16x32 f16) * B(32x16 f16) + C(16x16 f32)
  return __builtin_amdgcn_wmma_f32_16x16x32_f16(false, a, false, b, (short)0, c, false, false);
}
// A-operand: lane(0-15)=row M, lane>>4 selects K half; 16 contiguous K halves
// [base..base+15] -> e0..15.   base = row*stride + kbase + kg*16
__device__ __forceinline__ v16h loadA16(const _Float16* p) {
  V16U u; u.h[0] = *(const v8h*)p; u.h[1] = *(const v8h*)(p + 8); return u.v;
}
// B-operand: lane%16 = col N, lane>>4 selects K half; 16 contiguous K halves.
__device__ __forceinline__ v16h loadB16(const _Float16* p) {
  V16U u; u.h[0] = *(const v8h*)p; u.h[1] = *(const v8h*)(p + 8); return u.v;
}
__device__ __forceinline__ float siluf(float y) { return y / (1.f + expf(-y)); }

// CDNA5 async global->LDS copy (ASYNCcnt-tracked, GV addressing).
// lds_off: byte offset into this workgroup's LDS segment (dynamic base == 0).
__device__ __forceinline__ void async_copy_b128(unsigned lds_off, const void* gptr) {
  asm volatile("global_load_async_to_lds_b128 %0, %1, off"
               :: "v"(lds_off), "v"((unsigned long long)(uintptr_t)gptr)
               : "memory");
}
__device__ __forceinline__ void wait_async0() {
  asm volatile("s_wait_asynccnt 0x0" ::: "memory");
}

// ---------------------------------------------------------------- weights -> f16
__global__ void wcast_kernel(const float* __restrict__ w0, const float* __restrict__ w1,
                             const float* __restrict__ w2, const float* __restrict__ w3,
                             _Float16* __restrict__ out) {
  int idx = blockIdx.x * 256 + threadIdx.x;          // 4*512*512
  int z = idx >> 18; int i = idx & 262143;
  const float* s = (z == 0) ? w0 : (z == 1) ? w1 : (z == 2) ? w2 : w3;
  out[idx] = (_Float16)s[i];
}

// ---------------------------------------------------------------- x -> f16 (read 12x by proj)
__global__ void xcast_kernel(const float* __restrict__ x, _Float16* __restrict__ xh) {
  int idx = blockIdx.x * 256 + threadIdx.x;          // BL*D/4
  v4f v = *(const v4f*)(x + (size_t)idx * 4);
  v4h h;
#pragma unroll
  for (int j = 0; j < 4; ++j) h[j] = (_Float16)v[j];
  *(v4h*)(xh + (size_t)idx * 4) = h;
}

// ---------------------------------------------------------------- QKV projections (WMMA)
// out[r,n] = sum_d x[r,d] * W[n,d].  Weight panel (128 rows x 512) async-staged in LDS,
// shared by all 8 waves; B reads are conflict-free ds_load_b128 (row pitch 520 halves).
__global__ void __launch_bounds__(256) proj_kernel(const _Float16* __restrict__ xh,
                                                   const _Float16* __restrict__ wh,
                                                   _Float16* __restrict__ dst) {
  extern __shared__ char smem[];
  _Float16* Bp = (_Float16*)smem;                    // [128][520] f16
  int tid = threadIdx.x;
  int lane = tid & 31, wv = tid >> 5;
  int lhi = lane >> 4, llo = lane & 15;
  const _Float16* w = wh + (size_t)blockIdx.z * D_ * D_;
  _Float16* out = dst + (size_t)blockIdx.z * BL_ * D_;
  int m0 = blockIdx.x * 128 + wv * 16;
  int n0 = blockIdx.y * 128;

  // async-stage weight panel: 128 rows * 1KB = 8192 x16B units
#pragma unroll
  for (int it = 0; it < 32; ++it) {
    int e = it * 256 + tid;
    int rrow = e >> 6, rcol = e & 63;
    async_copy_b128((unsigned)((rrow * 520 + rcol * 8) * 2),
                    w + (size_t)(n0 + rrow) * D_ + rcol * 8);
  }
  wait_async0();
  __syncthreads();

  v8f acc[8];
#pragma unroll
  for (int i = 0; i < 8; ++i)
#pragma unroll
    for (int r = 0; r < 8; ++r) acc[i][r] = 0.f;

  int row = m0 + llo;
  for (int k = 0; k < D_; k += 32) {
    v16h av = loadA16(xh + (size_t)row * D_ + k + lhi * 16);
    v16h bv[8];
#pragma unroll
    for (int i = 0; i < 8; ++i)
      bv[i] = loadB16(Bp + (i * 16 + llo) * 520 + k + lhi * 16);
#pragma unroll
    for (int i = 0; i < 8; ++i) acc[i] = wmma16(av, bv[i], acc[i]);
  }
#pragma unroll
  for (int i = 0; i < 8; ++i)
#pragma unroll
    for (int r = 0; r < 8; ++r) {
      int m = m0 + r + 8 * lhi;
      int n = n0 + i * 16 + llo;
      out[(size_t)m * D_ + n] = (_Float16)acc[i][r];
    }
}

// ---------------------------------------------------------------- gate rank-16
__global__ void gate1_kernel(const float* __restrict__ x, const float* __restrict__ Wg1,
                             float* __restrict__ g1) {
  int t = threadIdx.x;
  int r = blockIdx.x * 16 + (t >> 4);
  int j = t & 15;
  const float* xr = x + (size_t)r * D_;
  const float* wr = Wg1 + (size_t)j * D_;
  float s = 0.f;
  for (int d = 0; d < D_; d += 4) {
    v4f xv = *(const v4f*)(xr + d);
    v4f wv = *(const v4f*)(wr + d);
    s += xv[0] * wv[0] + xv[1] * wv[1] + xv[2] * wv[2] + xv[3] * wv[3];
  }
  g1[r * 16 + j] = s;
}

__global__ void alpha_kernel(const float* __restrict__ g1, const float* __restrict__ Wg2,
                             const float* __restrict__ bg2, _Float16* __restrict__ al) {
  int idx = blockIdx.x * 256 + threadIdx.x;          // BL*D
  int r = idx >> 9, d = idx & 511;
  float s = bg2[d];
  const float* g = g1 + r * 16;
  const float* w = Wg2 + d * 16;
#pragma unroll
  for (int j = 0; j < 16; ++j) s += g[j] * w[j];
  float ls = (s <= 0.f) ? (s - log1pf(expf(s))) : (-log1pf(expf(-s)));
  al[idx] = (_Float16)(ls * (1.f / 16.f));
}

// ---------------------------------------------------------------- conv(K=4)+silu; v -> V^T layout
__global__ void conv_kernel(const _Float16* __restrict__ pk, const _Float16* __restrict__ pq,
                            const _Float16* __restrict__ pv,
                            const float* __restrict__ ck, const float* __restrict__ cq,
                            const float* __restrict__ cv,
                            _Float16* __restrict__ ksb, _Float16* __restrict__ qsb,
                            _Float16* __restrict__ VT) {
  int idx = blockIdx.x * 256 + threadIdx.x;          // BL*D
  int r = idx >> 9, d = idx & 511;
  int t = r & (L_ - 1);
  float yk = 0.f, yq = 0.f, yv = 0.f;
#pragma unroll
  for (int j = 0; j < 4; ++j) {
    int tt = t + j - 3;
    if (tt >= 0) {
      size_t ri = (size_t)(r + j - 3) * D_ + d;
      yk += (float)pk[ri] * ck[d * 4 + j];
      yq += (float)pq[ri] * cq[d * 4 + j];
      yv += (float)pv[ri] * cv[d * 4 + j];
    }
  }
  ksb[idx] = (_Float16)siluf(yk);
  qsb[idx] = (_Float16)siluf(yq);
  int b = r >> 14, n = t >> 6, tc = t & 63;
  VT[(((size_t)(b * NCH_ + n) * D_ + d) << 6) + tc] = (_Float16)siluf(yv);
}

// ---------------------------------------------------------------- q/k row L2 norms (wave per row)
__global__ void norm_kernel(const _Float16* __restrict__ qsb, const _Float16* __restrict__ ksb,
                            float* __restrict__ rnq, float* __restrict__ rnk) {
  int lane = threadIdx.x & 31, wv = threadIdx.x >> 5;
  int r = blockIdx.x * 8 + wv;
  {
    const _Float16* p = qsb + (size_t)r * D_ + lane * 16;
    v8h a = *(const v8h*)p, b = *(const v8h*)(p + 8);
    float s = 0.f;
#pragma unroll
    for (int j = 0; j < 8; ++j) { float u = (float)a[j], w = (float)b[j]; s += u * u + w * w; }
    for (int off = 16; off; off >>= 1) s += __shfl_xor(s, off, 32);
    if (lane == 0) rnq[r] = 1.f / (sqrtf(s) + 1e-5f);
  }
  {
    const _Float16* p = ksb + (size_t)r * D_ + lane * 16;
    v8h a = *(const v8h*)p, b = *(const v8h*)(p + 8);
    float s = 0.f;
#pragma unroll
    for (int j = 0; j < 8; ++j) { float u = (float)a[j], w = (float)b[j]; s += u * u + w * w; }
    for (int off = 16; off; off >>= 1) s += __shfl_xor(s, off, 32);
    if (lane == 0) rnk[r] = 1.f / (sqrtf(s) + 1e-5f);
  }
}

// ---------------------------------------------------------------- decay precompute: Qe, Ke, Ku^T, e_end
__global__ void gla_prep_kernel(const _Float16* __restrict__ al, const _Float16* __restrict__ qsb,
                                const _Float16* __restrict__ ksb,
                                const float* __restrict__ rnq, const float* __restrict__ rnk,
                                _Float16* __restrict__ Qe, _Float16* __restrict__ Ke,
                                _Float16* __restrict__ KuT, float* __restrict__ eend) {
  int idx = blockIdx.x * 256 + threadIdx.x;          // B*NCH*D
  int d = idx & 511, n = (idx >> 9) & 255, b = idx >> 17;
  int rb = b * L_ + n * CH_;
  float end = 0.f;
  for (int t = 0; t < CH_; ++t) end += (float)al[(size_t)(rb + t) * D_ + d];
  float ca = 0.f;
  size_t kub = ((size_t)(b * NCH_ + n) * D_ + d) << 6;
  for (int t = 0; t < CH_; ++t) {
    int row = rb + t;
    ca += (float)al[(size_t)row * D_ + d];
    float qv = (float)qsb[(size_t)row * D_ + d] * rnq[row];
    float kv = (float)ksb[(size_t)row * D_ + d] * rnk[row];
    Qe[(size_t)row * D_ + d] = (_Float16)(qv * expf(ca));
    Ke[(size_t)row * D_ + d] = (_Float16)(kv * expf(-ca));
    KuT[kub + t] = (_Float16)(kv * expf(end - ca));
  }
  eend[idx] = expf(end);
}

// ---------------------------------------------------------------- W = tril(Qe @ Ke^T), chunk-parallel
// Ke chunk (64x512 f16) async-staged in LDS, shared by the 4 waves; A row block shared per wave.
__global__ void __launch_bounds__(128) wpass_kernel(const _Float16* __restrict__ Qe,
                                                    const _Float16* __restrict__ Ke,
                                                    _Float16* __restrict__ Wbuf) {
  extern __shared__ char smem[];
  _Float16* Kp = (_Float16*)smem;                    // [64][520]
  int tid = threadIdx.x;
  int lane = tid & 31, wv = tid >> 5;                // 4 waves
  int lhi = lane >> 4, llo = lane & 15;
  int cid = blockIdx.x;                              // 0..511
  int b = cid >> 8, n = cid & 255;
  int rb = b * L_ + n * CH_;

#pragma unroll
  for (int it = 0; it < 32; ++it) {
    int e = it * 128 + tid;                          // 64 rows * 64 x16B units
    int rrow = e >> 6, rcol = e & 63;
    async_copy_b128((unsigned)((rrow * 520 + rcol * 8) * 2),
                    Ke + (size_t)(rb + rrow) * D_ + rcol * 8);
  }
  wait_async0();
  __syncthreads();

  _Float16* wout = Wbuf + (size_t)cid * 64 * 64;
  int tm = wv;                                       // A row block shared by this wave's 4 tiles
  v8f acc[4];
#pragma unroll
  for (int i = 0; i < 4; ++i)
#pragma unroll
    for (int r = 0; r < 8; ++r) acc[i][r] = 0.f;

  for (int k = 0; k < D_; k += 32) {
    v16h av = loadA16(Qe + (size_t)(rb + tm * 16 + llo) * D_ + k + lhi * 16);
    v16h bv[4];
#pragma unroll
    for (int tn = 0; tn < 4; ++tn)
      bv[tn] = loadB16(Kp + (tn * 16 + llo) * 520 + k + lhi * 16);
#pragma unroll
    for (int tn = 0; tn < 4; ++tn) acc[tn] = wmma16(av, bv[tn], acc[tn]);
  }
#pragma unroll
  for (int tn = 0; tn < 4; ++tn)
#pragma unroll
    for (int r = 0; r < 8; ++r) {
      int m = tm * 16 + r + 8 * lhi;
      int c = tn * 16 + llo;
      wout[m * 64 + c] = (_Float16)((c <= m) ? acc[tn][r] : 0.f);
    }
}

// ---------------------------------------------------------------- sequential scan, S resident in LDS
// grid (8 colgroups, 2 batches), 256 threads. dynamic LDS:
//   S32  [512][64] f32          131072 B
//   SfT  [64][520] f16 (S^T)     66560 B   (pad 8 halves/row -> conflict-free b128)
//   eendS[512] f32                2048 B
__global__ void __launch_bounds__(256) gla_scan_kernel(const _Float16* __restrict__ Qe,
                                                       const _Float16* __restrict__ Wbuf,
                                                       const _Float16* __restrict__ VT,
                                                       const _Float16* __restrict__ KuT,
                                                       const float* __restrict__ eend,
                                                       float* __restrict__ obuf) {
  extern __shared__ char smem[];
  float*    S32   = (float*)smem;
  _Float16* SfT   = (_Float16*)(smem + 512 * 64 * 4);
  float*    eendS = (float*)(smem + 512 * 64 * 4 + 64 * 520 * 2);
  const int SFP = 520;

  int tid = threadIdx.x, lane = tid & 31, wv = tid >> 5;
  int lhi = lane >> 4, llo = lane & 15;
  int col0 = blockIdx.x * 64;
  int b = blockIdx.y;

  for (int i = tid; i < 512 * 64; i += 256) S32[i] = 0.f;
  for (int i = tid; i < 64 * SFP; i += 256) SfT[i] = (_Float16)0.f;
  __syncthreads();

  // phase-a tile map: 2 output tiles per wave sharing the A operand
  int tm  = wv >> 1;
  int tn0 = (wv & 1) * 2;
  int tn1 = tn0 + 1;

  for (int n = 0; n < NCH_; ++n) {
    size_t cb = (size_t)(b * NCH_ + n);
    for (int i = tid; i < 512; i += 256) eendS[i] = eend[cb * 512 + i];
    int rb = b * L_ + n * CH_;
    const _Float16* vtc = VT + (cb * D_) * 64;

    // preload the 8 V^T B-operands (4 col-tiles x 2 K-halves): reused by phase a (W@V)
    // and by all 16 phase-c tiles.
    v16h vtB[4][2];
#pragma unroll
    for (int tn = 0; tn < 4; ++tn)
#pragma unroll
      for (int kk = 0; kk < 2; ++kk)
        vtB[tn][kk] = loadB16(vtc + (size_t)(col0 + tn * 16 + llo) * 64 + kk * 32 + lhi * 16);

    // ---- phase a: O = Qe @ S  +  W @ V  (this group's 64 cols)
    v8f a0, a1;
#pragma unroll
    for (int r = 0; r < 8; ++r) { a0[r] = 0.f; a1[r] = 0.f; }
    for (int k = 0; k < D_; k += 32) {
      v16h av = loadA16(Qe + (size_t)(rb + tm * 16 + llo) * D_ + k + lhi * 16);
      v16h b0 = loadB16(SfT + (tn0 * 16 + llo) * SFP + k + lhi * 16);
      v16h b1 = loadB16(SfT + (tn1 * 16 + llo) * SFP + k + lhi * 16);
      a0 = wmma16(av, b0, a0);
      a1 = wmma16(av, b1, a1);
    }
    {
      const _Float16* wr0 = Wbuf + cb * 4096 + (tm * 16 + llo) * 64 + lhi * 16;
      v16h aw0 = loadA16(wr0);
      v16h aw1 = loadA16(wr0 + 32);
      a0 = wmma16(aw0, vtB[tn0][0], a0);
      a0 = wmma16(aw1, vtB[tn0][1], a0);
      a1 = wmma16(aw0, vtB[tn1][0], a1);
      a1 = wmma16(aw1, vtB[tn1][1], a1);
    }
#pragma unroll
    for (int r = 0; r < 8; ++r) {
      int row = rb + tm * 16 + r + 8 * lhi;
      obuf[(size_t)row * D_ + col0 + tn0 * 16 + llo] = a0[r];
      obuf[(size_t)row * D_ + col0 + tn1 * 16 + llo] = a1[r];
    }
    __syncthreads();

    // ---- phase c: S = diag(e_end) * S + Ku^T @ V   (128 tiles, 16 per wave)
    const _Float16* kut = KuT + (cb * D_) * 64;
#pragma unroll 1
    for (int i = 0; i < 16; ++i) {
      int id = wv * 16 + i, tmu = id >> 2, tnu = id & 3;
      int nloc = tnu * 16 + llo;
      int d1b = tmu * 16 + 8 * lhi;
      const _Float16* ka = kut + (size_t)(tmu * 16 + llo) * 64 + lhi * 16;
      v16h ka0 = loadA16(ka);
      v16h ka1 = loadA16(ka + 32);
      v8f c;
#pragma unroll
      for (int r = 0; r < 8; ++r) c[r] = eendS[d1b + r] * S32[(d1b + r) * 64 + nloc];
      c = wmma16(ka0, vtB[tnu][0], c);
      c = wmma16(ka1, vtB[tnu][1], c);
      v8h p8;
#pragma unroll
      for (int r = 0; r < 8; ++r) { S32[(d1b + r) * 64 + nloc] = c[r]; p8[r] = (_Float16)c[r]; }
      *(v8h*)(SfT + nloc * SFP + d1b) = p8;
    }
    __syncthreads();
  }
}

// ---------------------------------------------------------------- RMS norm (wave per row)
__global__ void onorm_kernel(const float* __restrict__ obuf, const float* __restrict__ nw,
                             _Float16* __restrict__ onorm) {
  int lane = threadIdx.x & 31, wv = threadIdx.x >> 5;
  int r = blockIdx.x * 8 + wv;
  const float* orow = obuf + (size_t)r * D_;
  v4f vals[4];
  float s = 0.f;
#pragma unroll
  for (int j = 0; j < 4; ++j) {
    vals[j] = *(const v4f*)(orow + lane * 16 + j * 4);
    s += vals[j][0] * vals[j][0] + vals[j][1] * vals[j][1] +
         vals[j][2] * vals[j][2] + vals[j][3] * vals[j][3];
  }
  for (int off = 16; off; off >>= 1) s += __shfl_xor(s, off, 32);
  float rms = rsqrtf(s * (1.f / 512.f) + 1e-6f);
#pragma unroll
  for (int j = 0; j < 4; ++j)
#pragma unroll
    for (int e = 0; e < 4; ++e) {
      int d = lane * 16 + j * 4 + e;
      onorm[(size_t)r * D_ + d] = (_Float16)(vals[j][e] * rms * nw[d]);
    }
}

// ---------------------------------------------------------------- output GEMM (WMMA, f32 out)
__global__ void __launch_bounds__(256) outgemm_kernel(const _Float16* __restrict__ onorm,
                                                      const _Float16* __restrict__ wo,
                                                      float* __restrict__ out) {
  extern __shared__ char smem[];
  _Float16* Bp = (_Float16*)smem;                    // [128][520]
  int tid = threadIdx.x;
  int lane = tid & 31, wv = tid >> 5;
  int lhi = lane >> 4, llo = lane & 15;
  int m0 = blockIdx.x * 128 + wv * 16;
  int n0 = blockIdx.y * 128;

#pragma unroll
  for (int it = 0; it < 32; ++it) {
    int e = it * 256 + tid;
    int rrow = e >> 6, rcol = e & 63;
    async_copy_b128((unsigned)((rrow * 520 + rcol * 8) * 2),
                    wo + (size_t)(n0 + rrow) * D_ + rcol * 8);
  }
  wait_async0();
  __syncthreads();

  v8f acc[8];
#pragma unroll
  for (int i = 0; i < 8; ++i)
#pragma unroll
    for (int r = 0; r < 8; ++r) acc[i][r] = 0.f;
  int row = m0 + llo;
  for (int k = 0; k < D_; k += 32) {
    v16h av = loadA16(onorm + (size_t)row * D_ + k + lhi * 16);
    v16h bv[8];
#pragma unroll
    for (int i = 0; i < 8; ++i)
      bv[i] = loadB16(Bp + (i * 16 + llo) * 520 + k + lhi * 16);
#pragma unroll
    for (int i = 0; i < 8; ++i) acc[i] = wmma16(av, bv[i], acc[i]);
  }
#pragma unroll
  for (int i = 0; i < 8; ++i)
#pragma unroll
    for (int r = 0; r < 8; ++r) {
      int m = m0 + r + 8 * lhi;
      int n = n0 + i * 16 + llo;
      out[(size_t)m * D_ + n] = acc[i][r];
    }
}

// ================================================================ host
extern "C" void kernel_launch(void* const* d_in, const int* in_sizes, int n_in,
                              void* d_out, int out_size, void* d_ws, size_t ws_size,
                              hipStream_t stream) {
  const float* x   = (const float*)d_in[0];
  const float* Wk  = (const float*)d_in[1];
  const float* Wq  = (const float*)d_in[2];
  const float* Wv  = (const float*)d_in[3];
  const float* ck  = (const float*)d_in[4];
  const float* cq  = (const float*)d_in[5];
  const float* cv  = (const float*)d_in[6];
  const float* Wg1 = (const float*)d_in[7];
  const float* Wg2 = (const float*)d_in[8];
  const float* bg2 = (const float*)d_in[9];
  const float* nw  = (const float*)d_in[10];
  const float* Wo  = (const float*)d_in[11];
  char* ws = (char*)d_ws;

  constexpr size_t MB = 1024ull * 1024ull;
  constexpr size_t OFF_WH   = 0;                       //  2 MB f16: Wk,Wq,Wv,Wo
  constexpr size_t OFF_PK   = OFF_WH  + 2 * MB;        // 32 MB f16 proj-k
  constexpr size_t OFF_PQ   = OFF_PK  + 32 * MB;       // 32 MB f16 proj-q
  constexpr size_t OFF_PV   = OFF_PQ  + 32 * MB;       // 32 MB f16 proj-v
  constexpr size_t OFF_G1   = OFF_PV  + 32 * MB;       //  2 MB f32 gate-lowrank
  constexpr size_t OFF_AL   = OFF_G1  + 2 * MB;        // 32 MB f16 alpha
  constexpr size_t OFF_QS   = OFF_AL  + 32 * MB;       // 32 MB f16 silu(conv q)
  constexpr size_t OFF_KS   = OFF_QS  + 32 * MB;       // 32 MB f16 silu(conv k)
  constexpr size_t OFF_VT   = OFF_KS  + 32 * MB;       // 32 MB f16 V^T [b,n,d,t]
  constexpr size_t OFF_RNQ  = OFF_VT  + 32 * MB;       // 128 KB
  constexpr size_t OFF_RNK  = OFF_RNQ + 128 * 1024;    // 128 KB
  constexpr size_t OFF_QE   = OFF_RNK + 128 * 1024;    // 32 MB f16
  constexpr size_t OFF_KE   = OFF_QE  + 32 * MB;       // 32 MB f16
  constexpr size_t OFF_KUT  = OFF_KE  + 32 * MB;       // 32 MB f16 Ku^T [b,n,d,t]
  constexpr size_t OFF_EEND = OFF_KUT + 32 * MB;       //  1 MB f32
  // aliases over dead regions:
  constexpr size_t OFF_XH = OFF_VT;                    // 32 MB f16 x (dead before conv writes VT)
  constexpr size_t OFF_WB = OFF_QS;                    //  4 MB f16 tril(QK^T) per chunk
  constexpr size_t OFF_OB = OFF_PK;                    // 64 MB f32 scan output
  constexpr size_t OFF_ON = OFF_AL;                    // 32 MB f16 normalized o

  _Float16* wh   = (_Float16*)(ws + OFF_WH);
  _Float16* pk   = (_Float16*)(ws + OFF_PK);
  float*    g1   = (float*)   (ws + OFF_G1);
  _Float16* al   = (_Float16*)(ws + OFF_AL);
  _Float16* qsb  = (_Float16*)(ws + OFF_QS);
  _Float16* ksb  = (_Float16*)(ws + OFF_KS);
  _Float16* vt   = (_Float16*)(ws + OFF_VT);
  float*    rnq  = (float*)   (ws + OFF_RNQ);
  float*    rnk  = (float*)   (ws + OFF_RNK);
  _Float16* Qe   = (_Float16*)(ws + OFF_QE);
  _Float16* Ke   = (_Float16*)(ws + OFF_KE);
  _Float16* KuT  = (_Float16*)(ws + OFF_KUT);
  float*    eend = (float*)   (ws + OFF_EEND);
  _Float16* xh   = (_Float16*)(ws + OFF_XH);
  _Float16* wb   = (_Float16*)(ws + OFF_WB);
  float*    ob   = (float*)   (ws + OFF_OB);
  _Float16* on   = (_Float16*)(ws + OFF_ON);

  size_t panel_lds = 128 * 520 * 2;                    // 133120 B
  size_t kchunk_lds = 64 * 520 * 2;                    //  66560 B
  size_t scan_lds = 512 * 64 * 4 + 64 * 520 * 2 + 512 * 4;  // 199680 B

  wcast_kernel<<<4096, 256, 0, stream>>>(Wk, Wq, Wv, Wo, wh);
  xcast_kernel<<<16384, 256, 0, stream>>>(x, xh);
  proj_kernel<<<dim3(256, 4, 3), 256, panel_lds, stream>>>(xh, wh, pk);  // pk,pq,pv contiguous
  gate1_kernel<<<2048, 256, 0, stream>>>(x, Wg1, g1);
  alpha_kernel<<<65536, 256, 0, stream>>>(g1, Wg2, bg2, al);
  conv_kernel<<<65536, 256, 0, stream>>>(pk, pk + (size_t)BL_ * D_, pk + 2ull * BL_ * D_,
                                         ck, cq, cv, ksb, qsb, vt);
  norm_kernel<<<4096, 256, 0, stream>>>(qsb, ksb, rnq, rnk);
  gla_prep_kernel<<<1024, 256, 0, stream>>>(al, qsb, ksb, rnq, rnk, Qe, Ke, KuT, eend);
  wpass_kernel<<<512, 128, kchunk_lds, stream>>>(Qe, Ke, wb);
  gla_scan_kernel<<<dim3(8, 2), 256, scan_lds, stream>>>(Qe, wb, vt, KuT, eend, ob);
  onorm_kernel<<<4096, 256, 0, stream>>>(ob, nw, on);
  outgemm_kernel<<<dim3(256, 4), 256, panel_lds, stream>>>(on, wh + 3ull * D_ * D_, (float*)d_out);
}
